// MAUGCN_35373350650224
// MI455X (gfx1250) — matrix operand order repaired and checked by reading, not compiled, not run on hardware
//
#include <hip/hip_runtime.h>
#include <hip/hip_bf16.h>
#include <math.h>

// ---------------- CDNA5 vector types ----------------
typedef __attribute__((ext_vector_type(16))) _Float16 v16h;
typedef __attribute__((ext_vector_type(8)))  _Float16 v8h;
typedef __attribute__((ext_vector_type(8)))  float    v8f;
typedef __attribute__((ext_vector_type(4)))  int      v4i;

#define GM_AS  __attribute__((address_space(1)))
#define LDS_AS __attribute__((address_space(3)))

typedef GM_AS  v4i* gm_v4i_p;
typedef LDS_AS v4i* lds_v4i_p;

#if defined(__has_builtin)
#if __has_builtin(__builtin_amdgcn_global_load_async_to_lds_b128) && \
    __has_builtin(__builtin_amdgcn_s_wait_asynccnt)
#define HAVE_ASYNC_LDS 1
#endif
#endif
#ifndef HAVE_ASYNC_LDS
#define HAVE_ASYNC_LDS 0
#endif

// ---------------- problem constants (match reference) ----------------
static constexpr int   Nn    = 50000;
static constexpr int   Ee    = 800000;
static constexpr int   Vv    = 2;
static constexpr int   Fin   = 500;
static constexpr int   FinP  = 512;     // padded K for f16 B operand (16B-aligned rows)
static constexpr int   Cc    = 128;
static constexpr int   Ll    = 4;
static constexpr int   NCLS  = 16;
static constexpr int   KATT  = 5;
static constexpr float ALPHA = 0.1f;
static constexpr float LAMDA = 0.5f;

__device__ __forceinline__ float sigmoidf_(float x) { return 1.0f / (1.0f + expf(-x)); }

// =====================================================================
// WMMA GEMM:  D[M,NCOL] = A[M,K] (f32) * B[K,NCOL]
// B supplied transposed + zero-padded as Bt[NCOL][Kp] in f16 (Kp%32==0).
// Double-buffered LDS; B tiles staged with async-to-LDS (ASYNCcnt).
// EPI 0: D = relu(acc + bias[n])
// EPI 1: D = relu(tanh(theta*acc + (1-theta)*A[m,n])) * (watt? watt[m,n]:1)
// EPI 2: D = acc + bias[n]
// Block: 256 threads (8 waves), tile 128 x NCOL, K-step 32.
// =====================================================================
template <int NB, int EPI>
__global__ __launch_bounds__(256)
void gemm_wmma_kernel(const float* __restrict__ A,
                      const _Float16* __restrict__ Bt,
                      const float* __restrict__ bias,
                      const _Float16* __restrict__ watt,
                      float* __restrict__ D,
                      float theta, int M, int K, int Kp)
{
    constexpr int NCOL = NB * 16;
    __shared__ _Float16 As[2][128][40];    // pad rows to 40 halves: conflict-free b128 frag loads
    __shared__ _Float16 Bs[2][NCOL][40];

    const int tid  = threadIdx.x;
    const int wave = tid >> 5;
    const int lane = tid & 31;
    const int rowblk = blockIdx.x * 128;

    v8f acc[NB];
#pragma unroll
    for (int b = 0; b < NB; ++b)
#pragma unroll
        for (int j = 0; j < 8; ++j) acc[b][j] = 0.0f;

    const int r       = tid >> 1;          // 0..127 (LDS A row this thread stages)
    const int colbase = (tid & 1) * 16;    // 0 or 16
    const int gr      = rowblk + r;        // global A row
    const bool rowOK  = gr < M;

    float ar[16];
    auto fetchA = [&](int kt) {
        if (rowOK && (kt + colbase + 16 <= K)) {
            const float4* p = reinterpret_cast<const float4*>(A + (size_t)gr * K + kt + colbase);
#pragma unroll
            for (int q = 0; q < 4; ++q) {
                float4 v = p[q];
                ar[q*4+0] = v.x; ar[q*4+1] = v.y; ar[q*4+2] = v.z; ar[q*4+3] = v.w;
            }
        } else {
#pragma unroll
            for (int j = 0; j < 16; ++j) {
                int k = kt + colbase + j;
                ar[j] = (rowOK && k < K) ? A[(size_t)gr * K + k] : 0.0f;
            }
        }
    };
    auto commitA = [&](int buf) {
#pragma unroll
        for (int j = 0; j < 16; ++j) As[buf][r][colbase + j] = (_Float16)ar[j];
    };
    auto stageB = [&](int buf, int kt) {
#if HAVE_ASYNC_LDS
        // async copy: NCOL rows x 64B, in 16B chunks, ASYNCcnt-tracked
        for (int c = tid; c < NCOL * 4; c += 256) {
            const int n = c >> 2, part = c & 3;
            __builtin_amdgcn_global_load_async_to_lds_b128(
                (gm_v4i_p)(Bt + (size_t)n * Kp + kt + part * 8),
                (lds_v4i_p)(&Bs[buf][n][part * 8]), 0, 0);
        }
#else
        if (tid < 2 * NCOL) {
            const int n = tid >> 1;
#pragma unroll
            for (int j = 0; j < 16; ++j)
                Bs[buf][n][colbase + j] = Bt[(size_t)n * Kp + kt + colbase + j];
        }
#endif
    };

    // ---- prologue: stage tile 0 ----
    fetchA(0);
    stageB(0, 0);
    commitA(0);
#if HAVE_ASYNC_LDS
    __builtin_amdgcn_s_wait_asynccnt(0);
#endif
    __syncthreads();

    int buf = 0;
    for (int kt = 0; kt < K; kt += 32) {
        const bool more = (kt + 32) < K;
        if (more) {           // start staging next tile into buf^1
            stageB(buf ^ 1, kt + 32);
            fetchA(kt + 32);
        }

        // ---- A fragment (ISA 16-bit A 16x32 layout) ----
        const int mrow = wave * 16 + (lane & 15);
        const int koff = (lane >> 4) * 8;
        v8h alo = *reinterpret_cast<const v8h*>(&As[buf][mrow][koff]);
        v8h ahi = *reinterpret_cast<const v8h*>(&As[buf][mrow][16 + koff]);
        v16h afrag = __builtin_shufflevector(alo, ahi, 0,1,2,3,4,5,6,7,8,9,10,11,12,13,14,15);

        // ---- pipelined B fragments + WMMA ----
        const int kb = (lane >> 4) * 8;
        v8h blo = *reinterpret_cast<const v8h*>(&Bs[buf][lane & 15][kb]);
        v8h bhi = *reinterpret_cast<const v8h*>(&Bs[buf][lane & 15][kb + 16]);
#pragma unroll
        for (int b = 0; b < NB; ++b) {
            v16h bfrag = __builtin_shufflevector(blo, bhi, 0,1,2,3,4,5,6,7,8,9,10,11,12,13,14,15);
            if (b + 1 < NB) {
                const int n = (b + 1) * 16 + (lane & 15);
                blo = *reinterpret_cast<const v8h*>(&Bs[buf][n][kb]);
                bhi = *reinterpret_cast<const v8h*>(&Bs[buf][n][kb + 16]);
            }
            acc[b] = __builtin_amdgcn_wmma_f32_16x16x32_f16(
                false, afrag, false, bfrag, (short)0, acc[b], false, false);
        }

        if (more) {
            commitA(buf ^ 1);
#if HAVE_ASYNC_LDS
            __builtin_amdgcn_s_wait_asynccnt(0);
#endif
        }
        __syncthreads();
        buf ^= 1;
    }

    // ---- epilogue (C/D layout: VGPR rr -> M = rr + 8*(lane>=16), N = lane&15) ----
    const float oneMt = 1.0f - theta;
#pragma unroll
    for (int b = 0; b < NB; ++b) {
        const int gn = b * 16 + (lane & 15);
#pragma unroll
        for (int rr = 0; rr < 8; ++rr) {
            const int gm = rowblk + wave * 16 + rr + ((lane >> 4) << 3);
            if (gm < M) {
                float v = acc[b][rr];
                if (EPI == 0) {
                    v += bias[gn];
                    v = v > 0.0f ? v : 0.0f;
                } else if (EPI == 1) {
                    float s = A[(size_t)gm * K + gn];       // support (K==NCOL)
                    v = tanhf(theta * v + oneMt * s);
                    v = v > 0.0f ? v : 0.0f;
                    if (watt) v *= (float)watt[(size_t)gm * NCOL + gn];
                } else {
                    v += bias[gn];
                }
                D[(size_t)gm * NCOL + gn] = v;
            }
        }
    }
}

// =====================================================================
// SpMM scatter: out[row] += scale * val * h[col]  (one wave per edge,
// float4 gather of 128 channels; both operand and target are L2-resident
// (25.6MB each vs 192MB L2), atomics resolve at L2)
// =====================================================================
__global__ __launch_bounds__(256)
void spmm_atomic_kernel(const float* __restrict__ h,
                        const int* __restrict__ rows,
                        const int* __restrict__ cols,
                        const float* __restrict__ vals,
                        float* __restrict__ out,
                        int E, float scale)
{
    const int e = (blockIdx.x << 3) | (threadIdx.x >> 5);
    if (e >= E) return;
    const int lane = threadIdx.x & 31;
    const int rw = rows[e];
    const int cl = cols[e];
    const float v = vals[e] * scale;

    const float4 hv = reinterpret_cast<const float4*>(h + (size_t)cl * Cc)[lane];
    float* op = out + (size_t)rw * Cc + lane * 4;
    atomicAdd(op + 0, v * hv.x);
    atomicAdd(op + 1, v * hv.y);
    atomicAdd(op + 2, v * hv.z);
    atomicAdd(op + 3, v * hv.w);
}

// out[i] = a * in[i]
__global__ void scale_copy_kernel(const float* __restrict__ in, float* __restrict__ out,
                                  float a, size_t n)
{
    size_t i = (size_t)blockIdx.x * blockDim.x + threadIdx.x;
    size_t st = (size_t)gridDim.x * blockDim.x;
    for (; i < n; i += st) out[i] = a * in[i];
}

__global__ void cvt_f16_kernel(const float* __restrict__ in, _Float16* __restrict__ out, size_t n)
{
    size_t i = (size_t)blockIdx.x * blockDim.x + threadIdx.x;
    size_t st = (size_t)gridDim.x * blockDim.x;
    for (; i < n; i += st) out[i] = (_Float16)in[i];
}

// f32[rows][K] -> f16[rows][Kp], zero-padded columns K..Kp-1
__global__ void cvt_f16_pad_kernel(const float* __restrict__ in, _Float16* __restrict__ out,
                                   int rows, int K, int Kp)
{
    size_t total = (size_t)rows * Kp;
    size_t i = (size_t)blockIdx.x * blockDim.x + threadIdx.x;
    size_t st = (size_t)gridDim.x * blockDim.x;
    for (; i < total; i += st) {
        int n = (int)(i / Kp), k = (int)(i % Kp);
        out[i] = (k < K) ? (_Float16)in[(size_t)n * K + k] : (_Float16)0.0f;
    }
}

// =====================================================================
// Fused ECA-style attention (view 0): one block of 128 threads per node.
// =====================================================================
__global__ __launch_bounds__(128)
void attention_kernel(float* __restrict__ h,
                      const float* __restrict__ x2buf,
                      const float* __restrict__ cw,   // [5]
                      const float* __restrict__ mw,   // [1]
                      _Float16* __restrict__ watt,
                      int N)
{
    const int node = blockIdx.x;
    const int c    = threadIdx.x;           // 0..127
    const int wid  = c >> 5, lane = c & 31;

    __shared__ float xs[Cc + 4];
    __shared__ float os[Cc + 4];
    __shared__ float p1[4], p2[4];

    float w[KATT];
#pragma unroll
    for (int j = 0; j < KATT; ++j) w[j] = cw[j];

    const float xv  = h[(size_t)node * Cc + c];
    const float x2v = x2buf[(size_t)node * Cc + c];
    xs[c + 2] = xv;
    if (c < 2) { xs[c] = 0.0f; xs[Cc + 2 + c] = 0.0f; }
    __syncthreads();

    float x1 = 0.0f;
#pragma unroll
    for (int j = 0; j < KATT; ++j) x1 += w[j] * xs[c + j];

    float r1 = x1, r2 = x2v;
#pragma unroll
    for (int o = 16; o > 0; o >>= 1) {
        r1 += __shfl_xor(r1, o, 32);
        r2 += __shfl_xor(r2, o, 32);
    }
    if (lane == 0) { p1[wid] = r1; p2[wid] = r2; }
    __syncthreads();
    const float s1 = p1[0] + p1[1] + p1[2] + p1[3];
    const float s2 = p2[0] + p2[1] + p2[2] + p2[3];

    const float m    = sigmoidf_(mw[0]);
    const float out1 = sigmoidf_(s1 * x2v);
    const float out2 = sigmoidf_(s2 * x1);
    const float o    = m * out1 + (1.0f - m) * out2;

    os[c + 2] = o;
    if (c < 2) { os[c] = 0.0f; os[Cc + 2 + c] = 0.0f; }
    __syncthreads();

    float oc = 0.0f;
#pragma unroll
    for (int j = 0; j < KATT; ++j) oc += w[j] * os[c + j];
    oc = sigmoidf_(oc);

    watt[(size_t)node * Cc + c] = (_Float16)oc;
    h[(size_t)node * Cc + c]    = xv * oc;
}

// =====================================================================
// ortho_norm: per layer (one block, 128 threads), all in LDS (~131KB).
//   A = W.T@W + 1e-4 I ; L = chol(A) ; Y = inv(L) @ W.T  (Wo = Y.T)
// Emits Y[n][k] f16 — exactly the Bt layout gemm_wmma wants.
// =====================================================================
__global__ __launch_bounds__(128)
void ortho_kernel(const float* __restrict__ conv_w,   // [L][C][C]
                  _Float16* __restrict__ Y16)         // [L][C][C]
{
    const int l   = blockIdx.x;
    const int tid = threadIdx.x;
    __shared__ float Bm[Cc][Cc + 1];   // W.T, becomes Y
    __shared__ float Am[Cc][Cc];       // WtW, becomes chol L

    const float* W = conv_w + (size_t)l * Cc * Cc;

    for (int j = 0; j < Cc; ++j) Bm[j][tid] = W[(size_t)tid * Cc + j];
    __syncthreads();

    for (int rr = 0; rr < Cc; ++rr) {
        float s = 0.0f;
        for (int k = 0; k < Cc; ++k) s += Bm[rr][k] * Bm[tid][k];
        Am[rr][tid] = s + ((rr == tid) ? 1e-4f : 0.0f);
    }
    __syncthreads();

    for (int j = 0; j < Cc; ++j) {
        if (tid == j) Am[j][j] = sqrtf(Am[j][j]);
        __syncthreads();
        const float ljj = Am[j][j];
        if (tid > j) Am[tid][j] /= ljj;
        __syncthreads();
        if (tid > j) {
            const float lrj = Am[tid][j];
            for (int cidx = j + 1; cidx <= tid; ++cidx) Am[tid][cidx] -= lrj * Am[cidx][j];
        }
        __syncthreads();
    }

    for (int i = 0; i < Cc; ++i) {
        const float yi = Bm[i][tid] / Am[i][i];
        Bm[i][tid] = yi;
        for (int rr = i + 1; rr < Cc; ++rr) Bm[rr][tid] -= Am[rr][i] * yi;
    }
    __syncthreads();

    _Float16* Yo = Y16 + (size_t)l * Cc * Cc;
    for (int j = 0; j < Cc; ++j) Yo[(size_t)tid * Cc + j] = (_Float16)Bm[tid][j];
}

// =====================================================================
// Final: per-node 16-class log-softmax combinations -> d_out
// =====================================================================
__global__ void finalize_kernel(const float* __restrict__ lg0,
                                const float* __restrict__ lg1,
                                float* __restrict__ out, int N)
{
    int n = blockIdx.x * blockDim.x + threadIdx.x;
    if (n >= N) return;
    float a[NCLS], b[NCLS], t[NCLS], la[NCLS], lb[NCLS], lt[NCLS];
#pragma unroll
    for (int i = 0; i < NCLS; ++i) {
        a[i] = lg0[(size_t)n * NCLS + i];
        b[i] = lg1[(size_t)n * NCLS + i];
        t[i] = a[i] + b[i];
    }
    float ma = -1e30f, mb = -1e30f, mt = -1e30f;
#pragma unroll
    for (int i = 0; i < NCLS; ++i) { ma = fmaxf(ma, a[i]); mb = fmaxf(mb, b[i]); mt = fmaxf(mt, t[i]); }
    float sa = 0.0f, sb = 0.0f, st = 0.0f;
#pragma unroll
    for (int i = 0; i < NCLS; ++i) { sa += expf(a[i]-ma); sb += expf(b[i]-mb); st += expf(t[i]-mt); }
    const float lsa = logf(sa), lsb = logf(sb), lst = logf(st);
#pragma unroll
    for (int i = 0; i < NCLS; ++i) {
        la[i] = a[i] - ma - lsa;
        lb[i] = b[i] - mb - lsb;
        lt[i] = t[i] - mt - lst;
    }
    const size_t NE = (size_t)N * NCLS;
#pragma unroll
    for (int i = 0; i < NCLS; ++i) {
        out[0 * NE + (size_t)n * NCLS + i] = lt[i];
        out[1 * NE + (size_t)n * NCLS + i] = 0.5f * (la[i] + lb[i]);
        out[2 * NE + (size_t)n * NCLS + i] = la[i];
        out[3 * NE + (size_t)n * NCLS + i] = lb[i];
    }
}

// =====================================================================
// host launcher
// =====================================================================
extern "C" void kernel_launch(void* const* d_in, const int* in_sizes, int n_in,
                              void* d_out, int out_size, void* d_ws, size_t ws_size,
                              hipStream_t stream)
{
    const float* x          = (const float*)d_in[0];
    const int*   rows       = (const int*)  d_in[1];
    const int*   cols       = (const int*)  d_in[2];
    const float* vals       = (const float*)d_in[3];
    const float* fc_w       = (const float*)d_in[4];
    const float* fc_b       = (const float*)d_in[5];
    const float* conv_w     = (const float*)d_in[6];
    const float* att_conv_w = (const float*)d_in[7];
    const float* att_fc_w   = (const float*)d_in[8];
    const float* att_fc_b   = (const float*)d_in[9];
    const float* att_mix_w  = (const float*)d_in[10];
    const float* out_w      = (const float*)d_in[11];
    const float* out_b      = (const float*)d_in[12];
    float* out = (float*)d_out;

    // ---- workspace carve-up ----
    char* base = (char*)d_ws;
    size_t off = 0;
    auto alloc = [&](size_t bytes) -> char* {
        char* p = base + off;
        off = (off + bytes + 255) & ~(size_t)255;
        return p;
    };
    float*     h0     = (float*)    alloc((size_t)Nn * Cc * 4);
    float*     h      = (float*)    alloc((size_t)Nn * Cc * 4);
    float*     sup    = (float*)    alloc((size_t)Nn * Cc * 4);
    float*     x2buf  = (float*)    alloc((size_t)Nn * Cc * 4);
    _Float16*  watt   = (_Float16*) alloc((size_t)Ll * Nn * Cc * 2);
    float*     lg0    = (float*)    alloc((size_t)Nn * NCLS * 4);
    float*     lg1    = (float*)    alloc((size_t)Nn * NCLS * 4);
    _Float16*  fch16  = (_Float16*) alloc((size_t)Vv * Cc * FinP * 2);   // padded K
    _Float16*  attf16 = (_Float16*) alloc((size_t)Ll * Cc * Cc * 2);
    _Float16*  outw16 = (_Float16*) alloc((size_t)NCLS * Cc * 2);
    _Float16*  Y16    = (_Float16*) alloc((size_t)Ll * Cc * Cc * 2);

    // ---- weight prep (f32 -> f16, B^T layout, FC padded to K=512) ----
    cvt_f16_pad_kernel<<<512, 256, 0, stream>>>(fc_w, fch16, Vv * Cc, Fin, FinP);
    cvt_f16_kernel<<<256, 256, 0, stream>>>(att_fc_w, attf16, (size_t)Ll * Cc * Cc);
    cvt_f16_kernel<<<8, 256, 0, stream>>>(out_w, outw16, (size_t)NCLS * Cc);

    // ---- orthonormalize conv weights, emit Wo^T (f16) per layer ----
    ortho_kernel<<<Ll, 128, 0, stream>>>(conv_w, Y16);

    const dim3 gemmGrid((Nn + 127) / 128);
    const dim3 gemmBlk(256);
    const int spmmGrid = (Ee + 7) / 8;
    const size_t hBytes = (size_t)Nn * Cc * 4;

    for (int k = 0; k < Vv; ++k) {
        // h0 = relu(x_k @ fc_w_k^T + fc_b_k)
        gemm_wmma_kernel<8, 0><<<gemmGrid, gemmBlk, 0, stream>>>(
            x + (size_t)k * Nn * Fin, fch16 + (size_t)k * Cc * FinP,
            fc_b + (size_t)k * Cc, nullptr, h0, 0.0f, Nn, Fin, FinP);
        (void)hipMemcpyAsync(h, h0, hBytes, hipMemcpyDeviceToDevice, stream);

        for (int i = 0; i < Ll; ++i) {
            if (k == 0) {
                gemm_wmma_kernel<8, 2><<<gemmGrid, gemmBlk, 0, stream>>>(
                    h, attf16 + (size_t)i * Cc * Cc, att_fc_b + (size_t)i * Cc,
                    nullptr, x2buf, 0.0f, Nn, Cc, Cc);
                attention_kernel<<<Nn, 128, 0, stream>>>(
                    h, x2buf, att_conv_w + (size_t)i * KATT, att_mix_w + i,
                    watt + (size_t)i * Nn * Cc, Nn);
            }
            // support = ALPHA*h0 ; support += (1-ALPHA) * A @ h   (fused scatter)
            scale_copy_kernel<<<2048, 256, 0, stream>>>(h0, sup, ALPHA, (size_t)Nn * Cc);
            spmm_atomic_kernel<<<spmmGrid, 256, 0, stream>>>(
                h, rows + (size_t)k * Ee, cols + (size_t)k * Ee, vals + (size_t)k * Ee,
                sup, Ee, 1.0f - ALPHA);
            // h = relu(tanh(theta*(sup@Wo) + (1-theta)*sup)) [* watt_i for view 1]
            const float theta = logf(LAMDA / (float)(i + 1) + 1.0f);
            gemm_wmma_kernel<8, 1><<<gemmGrid, gemmBlk, 0, stream>>>(
                sup, Y16 + (size_t)i * Cc * Cc, nullptr,
                (k == 1) ? (watt + (size_t)i * Nn * Cc) : nullptr,
                h, theta, Nn, Cc, Cc);
        }
        // logits_k = h @ out_w^T + out_b
        gemm_wmma_kernel<1, 2><<<gemmGrid, gemmBlk, 0, stream>>>(
            h, outw16, out_b, nullptr, (k == 0) ? lg0 : lg1, 0.0f, Nn, Cc, Cc);
    }

    finalize_kernel<<<(Nn + 255) / 256, 256, 0, stream>>>(lg0, lg1, out, Nn);
}
